// AttentionBlock_41480794144784
// MI455X (gfx1250) — matrix operand order repaired and checked by reading, not compiled, not run on hardware
//
#include <hip/hip_runtime.h>
#include <hip/hip_bf16.h>
#include <math.h>

// ---------------------------------------------------------------------------
// Problem constants (match reference)
// ---------------------------------------------------------------------------
constexpr int BB  = 2;
constexpr int NN  = 2048;
constexpr int DD  = 512;
constexpr int HH  = 8;
constexpr int HD  = 64;
constexpr int BN  = BB * NN;      // 4096 rows
constexpr int DFF = 2 * DD;       // 1024

// ---------------------------------------------------------------------------
// WMMA types / helpers (CDNA5 wave32, 16x16x32 bf16 -> f32)
// ---------------------------------------------------------------------------
typedef __attribute__((ext_vector_type(16))) __bf16 v16bf;
typedef __attribute__((ext_vector_type(8)))  float  v8f;

static __device__ inline v8f wmma_bf16(v16bf a, v16bf b, v8f c) {
    return __builtin_amdgcn_wmma_f32_16x16x32_bf16(
        false, a, false, b, (short)0, c, false, false);
}

// A-matrix 16x32 per-lane layout (ISA 7.12.2): lane holds K = half*8..+7 in
// vector elems 0..7 and K = 16+half*8..+7 in elems 8..15.
static __device__ inline v16bf load_afrag(const unsigned short* base, int half) {
    v16bf a;
    uint4* p = (uint4*)&a;
    p[0] = *(const uint4*)(base + half * 8);
    p[1] = *(const uint4*)(base + 16 + half * 8);
    return a;
}

// B-matrix 32x16 per-lane layout: lane (col) holds K = half*8..+7 in elems 0..7
// and K = 16+half*8..+7 in elems 8..15 (same as A).
static __device__ inline v16bf load_bfrag(const unsigned short* base, int half) {
    v16bf b;
    uint4* p = (uint4*)&b;
    p[0] = *(const uint4*)(base + half * 8);
    p[1] = *(const uint4*)(base + 16 + half * 8);
    return b;
}

// f32 -> bf16 round-to-nearest-even
static __device__ inline unsigned short f2bf(float f) {
    unsigned int u = __float_as_uint(f);
    if ((u & 0x7fffffffu) > 0x7f800000u) return (unsigned short)((u >> 16) | 0x40);
    unsigned int r = u + 0x7fffu + ((u >> 16) & 1u);
    return (unsigned short)(r >> 16);
}

static __device__ inline float half16_max(float v) {
    for (int m = 1; m < 16; m <<= 1) v = fmaxf(v, __shfl_xor(v, m, 32));
    return v;
}
static __device__ inline float half16_sum(float v) {
    for (int m = 1; m < 16; m <<= 1) v += __shfl_xor(v, m, 32);
    return v;
}

// CDNA5 async global->LDS copy (16B per lane, tracked by ASYNCcnt).
// lds_off = wave-relative LDS byte offset (low 32 bits of generic pointer).
static __device__ inline void async_ld16(unsigned lds_off, const void* g) {
    asm volatile("global_load_async_to_lds_b128 %0, %1, off"
                 :: "v"(lds_off), "v"((unsigned long long)(uintptr_t)g)
                 : "memory");
}
static __device__ inline void wait_async_le3() {
    asm volatile("s_wait_asynccnt 0x3" ::: "memory");
}

// ---------------------------------------------------------------------------
// f32 -> bf16 TRANSPOSED weight convert: in[K][NC] -> out[NC][K]
// ---------------------------------------------------------------------------
__global__ __launch_bounds__(256) void cvt_bf16_t_kernel(const float* __restrict__ in,
                                                         unsigned short* __restrict__ out,
                                                         int K, int NC) {
    int i = blockIdx.x * 256 + threadIdx.x;
    if (i < K * NC) {
        int k = i / NC, n = i % NC;
        out[(size_t)n * K + k] = f2bf(in[i]);
    }
}

// ---------------------------------------------------------------------------
// LayerNorm: one wave per row of 512, output bf16.  Block = 256 (8 rows).
// ---------------------------------------------------------------------------
__global__ __launch_bounds__(256) void ln_kernel(const float* __restrict__ x,
                                                 const float* __restrict__ g,
                                                 const float* __restrict__ beta,
                                                 unsigned short* __restrict__ out) {
    int wave = threadIdx.x >> 5, lane = threadIdx.x & 31;
    int row  = blockIdx.x * 8 + wave;
    const float* xr = x + (size_t)row * DD;

    float va[16];
    float s = 0.f, s2 = 0.f;
    #pragma unroll
    for (int i = 0; i < 16; ++i) {
        float v = xr[lane + i * 32];
        va[i] = v; s += v; s2 += v * v;
    }
    for (int m = 1; m < 32; m <<= 1) { s += __shfl_xor(s, m, 32); s2 += __shfl_xor(s2, m, 32); }
    float mean = s * (1.0f / DD);
    float var  = s2 * (1.0f / DD) - mean * mean;
    float rstd = rsqrtf(var + 1e-5f);

    unsigned short* orow = out + (size_t)row * DD;
    #pragma unroll
    for (int i = 0; i < 16; ++i) {
        int c = lane + i * 32;
        orow[c] = f2bf((va[i] - mean) * rstd * g[c] + beta[c]);
    }
}

// ---------------------------------------------------------------------------
// Tiled bf16 WMMA GEMM with async double-buffered LDS staging:
//   C[M,NC] = A[M,K] @ Wt^T + bias (+resid)(+gelu),  Wt is [NC][K] bf16.
// WG tile 128x64, 8 waves (4 M x 2 N), each wave 2x2 16x16 tiles.
// Per K-tile each wave issues 3 global_load_async_to_lds_b128; the next tile
// is issued before waiting on the current one (s_wait_asynccnt 3), so HBM/L2
// traffic overlaps the v_wmma stream.
// ---------------------------------------------------------------------------
template <bool GELU, bool RESID, bool OUTBF>
__global__ __launch_bounds__(256) void gemm_bf16_kernel(
    const unsigned short* __restrict__ A,    // [M,K] bf16 row-major
    const unsigned short* __restrict__ Wt,   // [NC,K] bf16 row-major (transposed weight)
    const float* __restrict__ bias,          // [NC]
    const float* __restrict__ resid,         // [M,NC] f32 or nullptr
    void* __restrict__ outv,
    int K, int NC) {
    __shared__ alignas(32) unsigned short ldsA[2][128 * 32];
    __shared__ alignas(32) unsigned short ldsBt[2][64 * 32];   // [n][k]

    const int tid  = threadIdx.x;
    const int wave = tid >> 5, lane = tid & 31;
    const int wm = wave & 3, wn = wave >> 2;
    const int half = lane >> 4, lr = lane & 15;
    const int n0 = blockIdx.x * 64;
    const int m0 = blockIdx.y * 128;

    const unsigned ldsA_off = (unsigned)(uintptr_t)(void*)&ldsA[0][0];
    const unsigned ldsB_off = (unsigned)(uintptr_t)(void*)&ldsBt[0][0];

    // chunk geometry (8 bf16 = 16B per async lane-op)
    const int eA0 = tid * 8;                 // A chunk 0 element id
    const int eA1 = (tid + 256) * 8;         // A chunk 1 element id
    const int rA0 = eA0 >> 5, kA0 = eA0 & 31;
    const int rA1 = eA1 >> 5, kA1 = eA1 & 31;
    const int eB  = tid * 8;
    const int nB  = eB >> 5,  kB  = eB & 31;

    auto issue_tile = [&](int k0, int buf) {
        unsigned aBase = ldsA_off + (unsigned)buf * (128 * 32 * 2);
        unsigned bBase = ldsB_off + (unsigned)buf * (64 * 32 * 2);
        async_ld16(aBase + eA0 * 2, &A[(size_t)(m0 + rA0) * K + k0 + kA0]);
        async_ld16(aBase + eA1 * 2, &A[(size_t)(m0 + rA1) * K + k0 + kA1]);
        async_ld16(bBase + eB * 2, &Wt[(size_t)(n0 + nB) * K + k0 + kB]);
    };

    v8f acc[2][2];
    #pragma unroll
    for (int i = 0; i < 2; ++i)
        #pragma unroll
        for (int j = 0; j < 2; ++j) acc[i][j] = (v8f)(0.0f);

    int cur = 0;
    issue_tile(0, 0);
    for (int k0 = 0; k0 < K; k0 += 32) {
        // issue next tile (wrap to tile 0 at the end: keeps wait-count uniform
        // and all addresses in-bounds; the extra tile is never consumed)
        int k0n = (k0 + 32 < K) ? (k0 + 32) : 0;
        issue_tile(k0n, cur ^ 1);
        wait_async_le3();        // current tile landed; next still in flight
        __syncthreads();

        const unsigned short* aT = &ldsA[cur][0];
        const unsigned short* bT = &ldsBt[cur][0];
        v16bf af[2], bf[2];
        #pragma unroll
        for (int rt = 0; rt < 2; ++rt)
            af[rt] = load_afrag(&aT[(wm * 32 + rt * 16 + lr) * 32], half);
        #pragma unroll
        for (int ct = 0; ct < 2; ++ct)
            bf[ct] = load_bfrag(&bT[(wn * 32 + ct * 16 + lr) * 32], half);
        #pragma unroll
        for (int rt = 0; rt < 2; ++rt)
            #pragma unroll
            for (int ct = 0; ct < 2; ++ct)
                acc[rt][ct] = wmma_bf16(af[rt], bf[ct], acc[rt][ct]);

        __syncthreads();         // everyone done reading buf[cur]
        cur ^= 1;
    }

    // Epilogue
    #pragma unroll
    for (int rt = 0; rt < 2; ++rt) {
        #pragma unroll
        for (int ct = 0; ct < 2; ++ct) {
            int col = n0 + wn * 32 + ct * 16 + lr;
            float bv = bias[col];
            #pragma unroll
            for (int r = 0; r < 8; ++r) {
                int row = m0 + wm * 32 + rt * 16 + half * 8 + r;
                float v = acc[rt][ct][r] + bv;
                if constexpr (GELU) v = 0.5f * v * (1.0f + erff(v * 0.70710678118654752f));
                if constexpr (RESID) v += resid[(size_t)row * NC + col];
                if constexpr (OUTBF)
                    ((unsigned short*)outv)[(size_t)row * NC + col] = f2bf(v);
                else
                    ((float*)outv)[(size_t)row * NC + col] = v;
            }
        }
    }
}

// ---------------------------------------------------------------------------
// Flash attention: grid (B*H, N/128).  WG: 128 query rows, 8 waves of 16
// rows; loop 64-key tiles; QK^T and P@V via WMMA; online softmax in f32.
// q/k/v stored as [B*N, D] bf16 (head h at column offset h*64).
// ---------------------------------------------------------------------------
__global__ __launch_bounds__(256) void attn_kernel(
    const unsigned short* __restrict__ qb,
    const unsigned short* __restrict__ kb,
    const unsigned short* __restrict__ vb,
    unsigned short* __restrict__ ob) {
    __shared__ alignas(32) unsigned short ldsK[64 * 64];    // [key][hd]
    __shared__ alignas(32) unsigned short ldsVt[64 * 64];   // [hd][key]
    __shared__ alignas(32) unsigned short ldsP[8 * 16 * 64];// per-wave [row][key]

    const int tid  = threadIdx.x;
    const int wave = tid >> 5, lane = tid & 31;
    const int half = lane >> 4, lr = lane & 15;
    const int bh = blockIdx.x;
    const int b  = bh / HH, h = bh % HH;
    const int m0 = blockIdx.y * 128;
    const size_t bN = (size_t)b * NN;
    const int h64 = h * HD;
    const float scale = 0.125f;   // 1/sqrt(64)

    const unsigned short* qrow = qb + ((bN + m0 + wave * 16 + lr) * DD + h64);
    v16bf qf[2];
    qf[0] = load_afrag(qrow, half);
    qf[1] = load_afrag(qrow + 32, half);

    v8f of[4];
    #pragma unroll
    for (int i = 0; i < 4; ++i) of[i] = (v8f)(0.0f);
    float mst[8], lst[8];
    #pragma unroll
    for (int r = 0; r < 8; ++r) { mst[r] = -1e30f; lst[r] = 0.0f; }

    unsigned short* pW = &ldsP[wave * 16 * 64];

    for (int kt = 0; kt < NN / 64; ++kt) {
        __syncthreads();
        #pragma unroll
        for (int it = 0; it < 2; ++it) {
            int c   = tid + it * 256;         // 0..511
            int key = c >> 3, hb8 = (c & 7) * 8;
            size_t src = (bN + kt * 64 + key) * DD + h64 + hb8;
            *(uint4*)&ldsK[key * 64 + hb8] = *(const uint4*)&kb[src];
            uint4 vv = *(const uint4*)&vb[src];
            const unsigned short* vs = (const unsigned short*)&vv;
            #pragma unroll
            for (int i = 0; i < 8; ++i) ldsVt[(hb8 + i) * 64 + key] = vs[i];
        }
        __syncthreads();

        // S = Q @ K^T  (per-wave 16 rows x 64 keys)
        v8f s[4];
        #pragma unroll
        for (int ct = 0; ct < 4; ++ct) {
            s[ct] = (v8f)(0.0f);
            #pragma unroll
            for (int kk = 0; kk < 2; ++kk) {
                v16bf bf = load_bfrag(&ldsK[(ct * 16 + lr) * 64 + kk * 32], half);
                s[ct] = wmma_bf16(qf[kk], bf, s[ct]);
            }
        }

        // Online softmax (rows live in 16-lane halves)
        float mnew[8], alpha[8];
        #pragma unroll
        for (int r = 0; r < 8; ++r) {
            float mx = -1e30f;
            #pragma unroll
            for (int ct = 0; ct < 4; ++ct) mx = fmaxf(mx, s[ct][r]);
            mx = half16_max(mx) * scale;
            mnew[r]  = fmaxf(mst[r], mx);
            alpha[r] = __expf(mst[r] - mnew[r]);
            mst[r]   = mnew[r];
        }
        #pragma unroll
        for (int r = 0; r < 8; ++r) {
            float rs = 0.0f;
            int prow = half * 8 + r;
            #pragma unroll
            for (int ct = 0; ct < 4; ++ct) {
                float p = __expf(s[ct][r] * scale - mnew[r]);
                pW[prow * 64 + ct * 16 + lr] = f2bf(p);
                rs += p;
            }
            rs = half16_sum(rs);
            lst[r] = lst[r] * alpha[r] + rs;
            #pragma unroll
            for (int nt = 0; nt < 4; ++nt) of[nt][r] *= alpha[r];
        }

        // O += P @ V   (per-wave; P scratch is wave-private)
        #pragma unroll
        for (int nt = 0; nt < 4; ++nt) {
            #pragma unroll
            for (int kk = 0; kk < 2; ++kk) {
                v16bf af = load_afrag(&pW[lr * 64 + kk * 32], half);
                v16bf bf = load_bfrag(&ldsVt[(nt * 16 + lr) * 64 + kk * 32], half);
                of[nt] = wmma_bf16(af, bf, of[nt]);
            }
        }
    }

    // Normalize and store bf16 into attn buffer [B*N, D]
    #pragma unroll
    for (int r = 0; r < 8; ++r) {
        float inv = 1.0f / lst[r];
        int grow = m0 + wave * 16 + half * 8 + r;
        #pragma unroll
        for (int nt = 0; nt < 4; ++nt) {
            int col = h64 + nt * 16 + lr;
            ob[(bN + grow) * DD + col] = f2bf(of[nt][r] * inv);
        }
    }
}

// ---------------------------------------------------------------------------
// Host launcher
// ---------------------------------------------------------------------------
extern "C" void kernel_launch(void* const* d_in, const int* in_sizes, int n_in,
                              void* d_out, int out_size, void* d_ws, size_t ws_size,
                              hipStream_t stream) {
    const float* y     = (const float*)d_in[0];
    const float* Wq    = (const float*)d_in[1];
    const float* bq    = (const float*)d_in[2];
    const float* Wk    = (const float*)d_in[3];
    const float* bk    = (const float*)d_in[4];
    const float* Wv    = (const float*)d_in[5];
    const float* bv    = (const float*)d_in[6];
    const float* Wo    = (const float*)d_in[7];
    const float* bo    = (const float*)d_in[8];
    const float* ln1g  = (const float*)d_in[9];
    const float* ln1b  = (const float*)d_in[10];
    const float* ln2g  = (const float*)d_in[11];
    const float* ln2b  = (const float*)d_in[12];
    const float* W1    = (const float*)d_in[13];
    const float* b1    = (const float*)d_in[14];
    const float* W2    = (const float*)d_in[15];
    const float* b2    = (const float*)d_in[16];
    float* out = (float*)d_out;

    char* ws = (char*)d_ws;
    constexpr size_t SZ_ACT  = (size_t)BN * DD * 2;        // 4 MiB bf16
    constexpr size_t SZ_WDD  = (size_t)DD * DD * 2;        // 512 KiB
    constexpr size_t SZ_WDF  = (size_t)DD * DFF * 2;       // 1 MiB
    unsigned short* xb  = (unsigned short*)(ws + 0);                 // LN1 out (reused as LN2 out)
    unsigned short* q   = (unsigned short*)(ws + 1 * SZ_ACT);
    unsigned short* k   = (unsigned short*)(ws + 2 * SZ_ACT);
    unsigned short* v   = (unsigned short*)(ws + 3 * SZ_ACT);
    unsigned short* att = (unsigned short*)(ws + 4 * SZ_ACT);
    char* wbase = ws + 5 * SZ_ACT;
    unsigned short* Wqt = (unsigned short*)(wbase + 0 * SZ_WDD);
    unsigned short* Wkt = (unsigned short*)(wbase + 1 * SZ_WDD);
    unsigned short* Wvt = (unsigned short*)(wbase + 2 * SZ_WDD);
    unsigned short* Wot = (unsigned short*)(wbase + 3 * SZ_WDD);
    unsigned short* W1t = (unsigned short*)(wbase + 4 * SZ_WDD);
    unsigned short* W2t = (unsigned short*)(wbase + 4 * SZ_WDD + SZ_WDF);
    float* y2 = (float*)(wbase + 4 * SZ_WDD + 2 * SZ_WDF);           // f32, 8 MiB
    unsigned short* hb  = xb;      // LN2 out reuses xb
    unsigned short* h1b = q;       // gelu(FFN1) reuses q+k region (8 MiB)

    // 1) weights -> transposed bf16 ([NC][K]) so GEMM B-tiles are straight copies
    cvt_bf16_t_kernel<<<(DD * DD) / 256, 256, 0, stream>>>(Wq, Wqt, DD, DD);
    cvt_bf16_t_kernel<<<(DD * DD) / 256, 256, 0, stream>>>(Wk, Wkt, DD, DD);
    cvt_bf16_t_kernel<<<(DD * DD) / 256, 256, 0, stream>>>(Wv, Wvt, DD, DD);
    cvt_bf16_t_kernel<<<(DD * DD) / 256, 256, 0, stream>>>(Wo, Wot, DD, DD);
    cvt_bf16_t_kernel<<<(DD * DFF) / 256, 256, 0, stream>>>(W1, W1t, DD, DFF);
    cvt_bf16_t_kernel<<<(DFF * DD) / 256, 256, 0, stream>>>(W2, W2t, DFF, DD);

    // 2) LN1
    ln_kernel<<<BN / 8, 256, 0, stream>>>(y, ln1g, ln1b, xb);

    // 3) QKV projections
    dim3 gDD(DD / 64, BN / 128);
    gemm_bf16_kernel<false, false, true><<<gDD, 256, 0, stream>>>(xb, Wqt, bq, nullptr, q, DD, DD);
    gemm_bf16_kernel<false, false, true><<<gDD, 256, 0, stream>>>(xb, Wkt, bk, nullptr, k, DD, DD);
    gemm_bf16_kernel<false, false, true><<<gDD, 256, 0, stream>>>(xb, Wvt, bv, nullptr, v, DD, DD);

    // 4) attention
    attn_kernel<<<dim3(BB * HH, NN / 128), 256, 0, stream>>>(q, k, v, att);

    // 5) output projection + residual: y2 = y + att @ Wo + bo
    gemm_bf16_kernel<false, true, false><<<gDD, 256, 0, stream>>>(att, Wot, bo, y, y2, DD, DD);

    // 6) LN2
    ln_kernel<<<BN / 8, 256, 0, stream>>>(y2, ln2g, ln2b, hb);

    // 7) FFN1 + exact GELU
    dim3 gDF(DFF / 64, BN / 128);
    gemm_bf16_kernel<true, false, true><<<gDF, 256, 0, stream>>>(hb, W1t, b1, nullptr, h1b, DD, DFF);

    // 8) FFN2 + residual -> out
    gemm_bf16_kernel<false, true, false><<<gDD, 256, 0, stream>>>(h1b, W2t, b2, y2, out, DFF, DD);

    (void)in_sizes; (void)n_in; (void)out_size; (void)ws_size;
}
